// SparseMoE_4569845203469
// MI455X (gfx1250) — matrix operand order repaired and checked by reading, not compiled, not run on hardware
//
#include <hip/hip_runtime.h>
#include <hip/hip_bf16.h>
#include <math.h>

// ---------------- problem constants (from reference) ----------------
#define Bb   4
#define Tt   4096
#define Dd   1024
#define Ee   8
#define Hh   4096
#define Nn   (Bb * Tt)      // 16384 tokens
#define CAP  4096           // N*TOP_K/E * capacity_factor
#define NB   64             // Nn / 256 tokens-per-block

typedef __bf16 bf16;
typedef __attribute__((ext_vector_type(16))) __bf16 v16bf;
typedef __attribute__((ext_vector_type(8)))  float  v8f;

union FragAB { v16bf v; uint4 u[2]; };

__device__ __forceinline__ bf16 f2bf(float f) {
  unsigned u = __builtin_bit_cast(unsigned, f);
  unsigned r = u + 0x7FFFu + ((u >> 16) & 1u);   // round-to-nearest-even
  unsigned short s = (unsigned short)(r >> 16);
  return __builtin_bit_cast(bf16, s);
}

// gfx1250 async copy: memory -> LDS, tracked by ASYNCcnt (no VGPR staging).
// Operand order per CDNA5 ISA: VDST = LDS byte address, VADDR = 64-bit global addr.
__device__ __forceinline__ void async_b128(unsigned ldsAddr, const bf16* g) {
  asm volatile("global_load_async_to_lds_b128 %0, %1, off"
               :: "v"(ldsAddr), "v"((unsigned long long)(uintptr_t)g)
               : "memory");
}

// ---------------- utility fills ----------------
__global__ void moe_fill_zero_f32(float* __restrict__ p, int n) {
  int i = blockIdx.x * blockDim.x + threadIdx.x;
  if (i < n) p[i] = 0.0f;
}
__global__ void moe_fill_zero_i32(int* __restrict__ p, int n) {
  int i = blockIdx.x * blockDim.x + threadIdx.x;
  if (i < n) p[i] = 0;
}

// ---------------- router: one wave (32 lanes) per token ----------------
__global__ void moe_router(const float* __restrict__ x, const float* __restrict__ noise_raw,
                           const float* __restrict__ Wr, const float* __restrict__ br,
                           const float* __restrict__ Wn, const float* __restrict__ bn,
                           int* __restrict__ top_idx, float* __restrict__ top_gate) {
  const int lane = threadIdx.x & 31;
  const int wv   = threadIdx.x >> 5;
  const int n    = blockIdx.x * 8 + wv;
  const float* xr = x + (size_t)n * Dd;

  float aR[Ee], aN[Ee];
#pragma unroll
  for (int e = 0; e < Ee; ++e) { aR[e] = 0.f; aN[e] = 0.f; }

  for (int d = lane; d < Dd; d += 32) {
    float xv = xr[d];
    const float4* wr4 = (const float4*)(Wr + (size_t)d * Ee);
    const float4* wn4 = (const float4*)(Wn + (size_t)d * Ee);
    float4 r0 = wr4[0], r1 = wr4[1];
    float4 n0 = wn4[0], n1 = wn4[1];
    aR[0] += xv * r0.x; aR[1] += xv * r0.y; aR[2] += xv * r0.z; aR[3] += xv * r0.w;
    aR[4] += xv * r1.x; aR[5] += xv * r1.y; aR[6] += xv * r1.z; aR[7] += xv * r1.w;
    aN[0] += xv * n0.x; aN[1] += xv * n0.y; aN[2] += xv * n0.z; aN[3] += xv * n0.w;
    aN[4] += xv * n1.x; aN[5] += xv * n1.y; aN[6] += xv * n1.z; aN[7] += xv * n1.w;
  }
#pragma unroll
  for (int e = 0; e < Ee; ++e) {
#pragma unroll
    for (int off = 16; off > 0; off >>= 1) {
      aR[e] += __shfl_xor(aR[e], off, 32);
      aN[e] += __shfl_xor(aN[e], off, 32);
    }
  }
  if (lane == 0) {
    float noisy[Ee];
#pragma unroll
    for (int e = 0; e < Ee; ++e) {
      float lg = aR[e] + br[e];
      float nl = aN[e] + bn[e];
      float sp = (nl > 20.0f) ? nl : log1pf(expf(nl));   // softplus
      noisy[e] = lg + noise_raw[(size_t)n * Ee + e] * sp;
    }
    int i0 = 0; float v0 = noisy[0];
#pragma unroll
    for (int e = 1; e < Ee; ++e) if (noisy[e] > v0) { v0 = noisy[e]; i0 = e; }
    int i1 = -1; float v1 = -3.0e38f;
#pragma unroll
    for (int e = 0; e < Ee; ++e) if (e != i0 && noisy[e] > v1) { v1 = noisy[e]; i1 = e; }
    float p1  = expf(v1 - v0);          // v0 >= v1
    float inv = 1.0f / (1.0f + p1);
    top_idx [2 * n]     = i0;
    top_idx [2 * n + 1] = i1;
    top_gate[2 * n]     = inv;
    top_gate[2 * n + 1] = p1 * inv;
  }
}

// ---------------- dispatch: per-block expert counts ----------------
__global__ void moe_count(const int* __restrict__ top_idx, int* __restrict__ bc) {
  __shared__ int cnt[Ee];
  int tid = threadIdx.x;
  if (tid < Ee) cnt[tid] = 0;
  __syncthreads();
  int n = blockIdx.x * 256 + tid;
  atomicAdd(&cnt[top_idx[2 * n]], 1);
  atomicAdd(&cnt[top_idx[2 * n + 1]], 1);
  __syncthreads();
  if (tid < Ee) bc[blockIdx.x * Ee + tid] = cnt[tid];
}

// exclusive prefix over blocks, one thread per expert
__global__ void moe_scan(const int* __restrict__ bc, int* __restrict__ bo) {
  int e = threadIdx.x;
  if (e < Ee) {
    int s = 0;
    for (int b = 0; b < NB; ++b) { bo[b * Ee + e] = s; s += bc[b * Ee + e]; }
  }
}

// stable flat-order slot assignment via ballot/popcount ranks
__global__ void moe_dispatch(const int* __restrict__ top_idx, const float* __restrict__ top_gate,
                             const int* __restrict__ bo,
                             int* __restrict__ token_ids, float* __restrict__ gates) {
  __shared__ int waveCnt[8][Ee];
  __shared__ int waveOff[8][Ee];
  const int tid = threadIdx.x, lane = tid & 31, w = tid >> 5;
  const int n = blockIdx.x * 256 + tid;
  const int e0 = top_idx[2 * n], e1 = top_idx[2 * n + 1];
  const float g0 = top_gate[2 * n], g1 = top_gate[2 * n + 1];
  const unsigned long long lt = (1ull << lane) - 1ull;
  int rank0 = 0, rank1 = 0;
#pragma unroll
  for (int e = 0; e < Ee; ++e) {
    unsigned long long bl = __ballot((e0 == e) || (e1 == e));
    if (lane == e) waveCnt[w][e] = __popcll(bl);
    if (e0 == e) rank0 = __popcll(bl & lt);
    if (e1 == e) rank1 = __popcll(bl & lt);
  }
  __syncthreads();
  if (tid < Ee) {
    int s = 0;
    for (int wv = 0; wv < 8; ++wv) { waveOff[wv][tid] = s; s += waveCnt[wv][tid]; }
  }
  __syncthreads();
  {
    int slot = bo[blockIdx.x * Ee + e0] + waveOff[w][e0] + rank0;
    if (slot < CAP) { token_ids[e0 * CAP + slot] = n; gates[e0 * CAP + slot] = g0; }
  }
  {
    int slot = bo[blockIdx.x * Ee + e1] + waveOff[w][e1] + rank1;
    if (slot < CAP) { token_ids[e1 * CAP + slot] = n; gates[e1 * CAP + slot] = g1; }
  }
}

// ---------------- gather tokens -> bf16 ----------------
__global__ void moe_gather(const float* __restrict__ x, const int* __restrict__ token_ids,
                           bf16* __restrict__ xg) {
  const int ec = blockIdx.x;              // e*CAP + c
  const int t  = token_ids[ec];
  const float* src = x + (size_t)t * Dd;
  bf16* dst = xg + (size_t)ec * Dd;
  for (int i = threadIdx.x; i < Dd; i += blockDim.x) dst[i] = f2bf(src[i]);
}

// ---------------- fp32 [R,C] -> bf16 transposed [C,R], per expert (z) ----------------
__global__ void moe_transpose_bf16(const float* __restrict__ in, bf16* __restrict__ out,
                                   int R, int C) {
  __shared__ float tile[32][33];
  const size_t eoff = (size_t)blockIdx.z * (size_t)R * (size_t)C;
  const float* ine = in + eoff;
  bf16* oute = out + eoff;
  const int c0 = blockIdx.x * 32, r0 = blockIdx.y * 32;
  for (int j = threadIdx.y; j < 32; j += 8)
    tile[j][threadIdx.x] = ine[(size_t)(r0 + j) * C + c0 + threadIdx.x];
  __syncthreads();
  for (int j = threadIdx.y; j < 32; j += 8)
    oute[(size_t)(c0 + j) * R + r0 + threadIdx.x] = f2bf(tile[threadIdx.x][j]);
}

// ---------------- WMMA GEMM: C[M,N] = A[M,K](bf16) x Bt[N,K](bf16)^T ----------------
// Double-buffered LDS; tiles fetched with global_load_async_to_lds_b128 (ASYNCcnt);
// branch-free unrolled-by-2 steady state (nk is always even), 2-tile peeled tail.
// RELU_BF16:  out = relu(C + bias[n]) -> bf16
// else:       v = (C + bias[n]) * gate[m]; atomicAdd(dst[tok[m]*ldDst + n], v)
template <bool RELU_BF16>
__global__ __launch_bounds__(256) void moe_gemm_wmma(
    const bf16* __restrict__ A, const bf16* __restrict__ Bm,
    int M, int N, int K, const float* __restrict__ bias,
    bf16* __restrict__ outBf,
    const int* __restrict__ tok, const float* __restrict__ gate,
    float* __restrict__ dst, int ldDst) {
  constexpr int LDSS = 40;                       // padded row stride (bf16 elems)
  __shared__ bf16 As[2][128 * LDSS];
  __shared__ bf16 Bs[2][128 * LDSS];

  const int lane  = threadIdx.x & 31;
  const int w     = threadIdx.x >> 5;
  const int waveM = (w & 3) * 32;                // 4 waves along M
  const int waveN = (w >> 2) * 64;               // 2 waves along N
  const int mBlock = blockIdx.y * 128;
  const int nBlock = blockIdx.x * 128;

  // loader layout: 256 threads x (2 rows of A + 2 rows of B), 16B per row-chunk.
  // Global pointers computed once; advanced by 32 elements per K-step (no muls in loop).
  const int lr = threadIdx.x >> 2;               // 0..63
  const int lc = (threadIdx.x & 3) * 8;          // 0,8,16,24 (bf16 elems)
  const bf16* gA0 = A  + (size_t)(mBlock + lr) * K + lc;
  const bf16* gA1 = gA0 + (size_t)64 * K;
  const bf16* gB0 = Bm + (size_t)(nBlock + lr) * K + lc;
  const bf16* gB1 = gB0 + (size_t)64 * K;

  const unsigned rowOff = (unsigned)((lr * LDSS + lc) * 2);
  const unsigned row64  = (unsigned)(64 * LDSS * 2);
  const unsigned ldsA[2] = { (unsigned)(uintptr_t)&As[0][0] + rowOff,
                             (unsigned)(uintptr_t)&As[1][0] + rowOff };
  const unsigned ldsB[2] = { (unsigned)(uintptr_t)&Bs[0][0] + rowOff,
                             (unsigned)(uintptr_t)&Bs[1][0] + rowOff };

  v8f c[2][4];
  const v8f zero = {0.f, 0.f, 0.f, 0.f, 0.f, 0.f, 0.f, 0.f};
#pragma unroll
  for (int i = 0; i < 2; ++i)
#pragma unroll
    for (int j = 0; j < 4; ++j) c[i][j] = zero;

  // issue one 128x32 A-tile + 128x32 B-tile (4 async instrs per wave, in order)
  auto issue = [&](int buf) {
    async_b128(ldsA[buf],         gA0);
    async_b128(ldsA[buf] + row64, gA1);
    async_b128(ldsB[buf],         gB0);
    async_b128(ldsB[buf] + row64, gB1);
    gA0 += 32; gA1 += 32; gB0 += 32; gB1 += 32;
  };

  // 8 WMMAs on one resident tile pair
  auto compute = [&](const bf16* AsR, const bf16* BsR) {
    FragAB a[2], b[4];
    // A frag (ISA 16-bit A 16x32): lanes 0-15 K in {0..7,16..23}, lanes 16-31 +8
    const int ar = lane & 15;
    const int ka = (lane < 16) ? 0 : 8;
#pragma unroll
    for (int i = 0; i < 2; ++i) {
      const bf16* p = &AsR[(waveM + i * 16 + ar) * LDSS + ka];
      a[i].u[0] = *(const uint4*)p;
      a[i].u[1] = *(const uint4*)(p + 16);
    }
    // B frag (ISA 16-bit B KxN): lanes 0-15 = col N, K in {0..7,16..23}; lanes 16-31 +8
    const int kb = (lane < 16) ? 0 : 8;
#pragma unroll
    for (int j = 0; j < 4; ++j) {
      const bf16* p = &BsR[(waveN + j * 16 + ar) * LDSS + kb];
      b[j].u[0] = *(const uint4*)p;
      b[j].u[1] = *(const uint4*)(p + 16);
    }
#pragma unroll
    for (int i = 0; i < 2; ++i)
#pragma unroll
      for (int j = 0; j < 4; ++j)
        c[i][j] = __builtin_amdgcn_wmma_f32_16x16x32_bf16(
            false, a[i].v, false, b[j].v, (short)0, c[i][j], false, false);
  };

  const int nk = K >> 5;                         // K/32, always even here
  issue(0);                                      // tile 0 -> buf0

  // steady state: two tiles per iteration, all buffer indices static
  for (int kt = 0; kt + 2 < nk; kt += 2) {
    issue(1);                                    // tile kt+1 -> buf1  (8 in flight)
    asm volatile("s_wait_asynccnt 0x4" ::: "memory");  // tile kt landed
    __syncthreads();
    compute(As[0], Bs[0]);
    __syncthreads();
    issue(0);                                    // tile kt+2 -> buf0
    asm volatile("s_wait_asynccnt 0x4" ::: "memory");  // tile kt+1 landed
    __syncthreads();
    compute(As[1], Bs[1]);
    __syncthreads();
  }
  // tail: tiles nk-2 (already in flight -> buf0) and nk-1
  issue(1);                                      // tile nk-1 -> buf1
  asm volatile("s_wait_asynccnt 0x4" ::: "memory");
  __syncthreads();
  compute(As[0], Bs[0]);
  __syncthreads();
  asm volatile("s_wait_asynccnt 0x0" ::: "memory");
  __syncthreads();
  compute(As[1], Bs[1]);

  // C/D layout: lane&15 = N col; VGPR r -> M = r (lanes 0-15) / 8+r (lanes 16-31)
  const int nL = lane & 15;
  const int mH = (lane < 16) ? 0 : 8;
#pragma unroll
  for (int i = 0; i < 2; ++i) {
#pragma unroll
    for (int j = 0; j < 4; ++j) {
      const int gn  = nBlock + waveN + j * 16 + nL;
      const float bv = bias[gn];
      const int gm0 = mBlock + waveM + i * 16 + mH;
#pragma unroll
      for (int r = 0; r < 8; ++r) {
        float v = c[i][j][r] + bv;
        if constexpr (RELU_BF16) {
          v = fmaxf(v, 0.0f);
          outBf[(size_t)(gm0 + r) * N + gn] = f2bf(v);
        } else {
          const int gm = gm0 + r;
          const float g = gate[gm];
          const int   t = tok[gm];
          atomicAdd(&dst[(size_t)t * ldDst + gn], v * g);
        }
      }
    }
  }
}

// ---------------- host orchestration ----------------
extern "C" void kernel_launch(void* const* d_in, const int* in_sizes, int n_in,
                              void* d_out, int out_size, void* d_ws, size_t ws_size,
                              hipStream_t stream) {
  const float* x         = (const float*)d_in[0];
  const float* noise_raw = (const float*)d_in[1];
  const float* Wr        = (const float*)d_in[2];
  const float* br        = (const float*)d_in[3];
  const float* Wn        = (const float*)d_in[4];
  const float* bn        = (const float*)d_in[5];
  const float* W1        = (const float*)d_in[6];
  const float* b1        = (const float*)d_in[7];
  const float* W2        = (const float*)d_in[8];
  const float* b2        = (const float*)d_in[9];
  float* out = (float*)d_out;

  char* ws = (char*)d_ws;
  size_t off = 0;
  auto alloc = [&](size_t bytes) -> void* {
    void* p = ws + off;
    off = (off + bytes + 255) & ~(size_t)255;
    return p;
  };
  bf16*  xg        = (bf16*)alloc((size_t)Ee * CAP * Dd * 2);   // 64 MB
  bf16*  w1t       = (bf16*)alloc((size_t)Ee * Hh * Dd * 2);    // 64 MB [E][H][D]
  bf16*  w2t       = (bf16*)alloc((size_t)Ee * Dd * Hh * 2);    // 64 MB [E][D][H]
  bf16*  h         = (bf16*)alloc((size_t)CAP * Hh * 2);        // 32 MB, reused per expert
  int*   token_ids = (int*)  alloc((size_t)Ee * CAP * 4);
  float* gates     = (float*)alloc((size_t)Ee * CAP * 4);
  int*   top_idx   = (int*)  alloc((size_t)Nn * 2 * 4);
  float* top_gate  = (float*)alloc((size_t)Nn * 2 * 4);
  int*   bc        = (int*)  alloc((size_t)NB * Ee * 4);
  int*   bo        = (int*)  alloc((size_t)NB * Ee * 4);

  // zero-init scatter target + dispatch tables (handles 0xAA poison)
  moe_fill_zero_f32<<<(Nn * Dd + 255) / 256, 256, 0, stream>>>(out, Nn * Dd);
  moe_fill_zero_f32<<<(Ee * CAP + 255) / 256, 256, 0, stream>>>(gates, Ee * CAP);
  moe_fill_zero_i32<<<(Ee * CAP + 255) / 256, 256, 0, stream>>>(token_ids, Ee * CAP);

  moe_router  <<<Nn / 8, 256, 0, stream>>>(x, noise_raw, Wr, br, Wn, bn, top_idx, top_gate);
  moe_count   <<<NB, 256, 0, stream>>>(top_idx, bc);
  moe_scan    <<<1, 32, 0, stream>>>(bc, bo);
  moe_dispatch<<<NB, 256, 0, stream>>>(top_idx, top_gate, bo, token_ids, gates);
  moe_gather  <<<Ee * CAP, 256, 0, stream>>>(x, token_ids, xg);

  dim3 tb(32, 8);
  moe_transpose_bf16<<<dim3(Hh / 32, Dd / 32, Ee), tb, 0, stream>>>(W1, w1t, Dd, Hh);
  moe_transpose_bf16<<<dim3(Dd / 32, Hh / 32, Ee), tb, 0, stream>>>(W2, w2t, Hh, Dd);

  for (int e = 0; e < Ee; ++e) {
    const bf16* xge = xg  + (size_t)e * CAP * Dd;
    const bf16* w1e = w1t + (size_t)e * Hh * Dd;
    const bf16* w2e = w2t + (size_t)e * Dd * Hh;
    // h = relu(xg @ W1 + b1)   [CAP x H]
    moe_gemm_wmma<true><<<dim3(Hh / 128, CAP / 128), 256, 0, stream>>>(
        xge, w1e, CAP, Hh, Dd, b1 + (size_t)e * Hh, h,
        nullptr, nullptr, nullptr, 0);
    // out += gate * (h @ W2 + b2) scattered by token id
    moe_gemm_wmma<false><<<dim3(Dd / 128, CAP / 128), 256, 0, stream>>>(
        h, w2e, CAP, Dd, Hh, b2 + (size_t)e * Dd, nullptr,
        token_ids + (size_t)e * CAP, gates + (size_t)e * CAP, out, Dd);
  }
}
